// EfficientSelfAttention_36051955483002
// MI455X (gfx1250) — matrix operand order repaired and checked by reading, not compiled, not run on hardware
//
#include <hip/hip_runtime.h>

// ---------------- problem constants ----------------
#define BATCH  4
#define NSEQ   4096
#define DMODEL 1024
#define NHEAD  16
#define DHEAD  64
// SCALE = sqrt(DHEAD) = 8 ; inv_qtr = 8^(-1/4) = 2^(-0.75)
#define INV_QTR 0.59460355750136051f

typedef __attribute__((ext_vector_type(16))) __bf16         bf16x16;
typedef __attribute__((ext_vector_type(8)))  float          f32x8;
typedef __attribute__((ext_vector_type(4)))  float          f32x4;
typedef __attribute__((ext_vector_type(8)))  unsigned short u16x8;
typedef __attribute__((ext_vector_type(4)))  int            i32x4;

#if defined(__gfx1250__) && __has_builtin(__builtin_amdgcn_global_load_async_to_lds_b128)
#define USE_ASYNC_LDS 1
typedef __attribute__((address_space(1))) i32x4 as1_i32x4;
typedef __attribute__((address_space(3))) i32x4 as3_i32x4;
#else
#define USE_ASYNC_LDS 0
#endif

__device__ __forceinline__ unsigned short f2bf(float f) {
  unsigned int u = __float_as_uint(f);
  u += 0x7FFFu + ((u >> 16) & 1u);          // round-to-nearest-even
  return (unsigned short)(u >> 16);
}
__device__ __forceinline__ float bf2f(unsigned short h) {
  return __uint_as_float(((unsigned int)h) << 16);
}

__device__ __forceinline__ bf16x16 pack_frag(u16x8 lo, u16x8 hi) {
  union { u16x8 h[2]; bf16x16 v; } u;
  u.h[0] = lo; u.h[1] = hi;
  return u.v;
}

// A fragment (16x32 bf16, MxK) from a row-major (K-contiguous) tile.
// ISA 7.12.2: lanes 0-15 hold M=lane, K {0..7,16..23}; lanes 16-31 hold
// M=lane-16, K {8..15,24..31}.
__device__ __forceinline__ bf16x16 frag_a(const unsigned short* A, int ld, int mbase, int lane) {
  const unsigned short* p = A + (mbase + (lane & 15)) * ld + ((lane & 16) ? 8 : 0);
  return pack_frag(*(const u16x8*)p, *(const u16x8*)(p + 16));
}
// B fragment (32x16 bf16, KxN) from a tile storing each output column as a
// K-contiguous row (i.e. B^T row-major). Lanes 0-15: col=lane, K {0..7,16..23};
// lanes 16-31: col=lane-16, K {8..15,24..31}.
__device__ __forceinline__ bf16x16 frag_b(const unsigned short* Bt, int ld, int nbase, int kbase, int lane) {
  const unsigned short* p = Bt + (nbase + (lane & 15)) * ld + kbase + ((lane & 16) ? 8 : 0);
  return pack_frag(*(const u16x8*)p, *(const u16x8*)(p + 16));
}

__device__ __forceinline__ f32x8 wmma_bf16(bf16x16 a, bf16x16 b, f32x8 c) {
  return __builtin_amdgcn_wmma_f32_16x16x32_bf16(false, a, false, b, (short)0, c, false, false);
}

// ================= Kernel 1: projection GEMM (fp32 in -> bf16 out) ==========
// out[b,h,n,dh] = sum_k x[b,n,k] * W[j,k] + bias[j],  j = h*64+dh
__global__ __launch_bounds__(256)
void proj_gemm_kernel(const float* __restrict__ x, const float* __restrict__ W,
                      const float* __restrict__ bias, unsigned short* __restrict__ out)
{
  constexpr int BM = 128, BN = 64, BK = 32, LDA = 40, LDB = 40;
  __shared__ unsigned short As[BM * LDA];
  __shared__ unsigned short Bs[BN * LDB];
  const int m0 = blockIdx.x * BM;
  const int n0 = blockIdx.y * BN;
  const int tid = threadIdx.x;
  const int lane = tid & 31;
  const int w = tid >> 5;
  const int wm = (w & 3) * 32;
  const int wn = (w >> 2) * 32;
  const int lr = tid >> 3;            // 0..31
  const int lc = (tid & 7) * 4;       // 0..28

  f32x8 c00 = {}, c01 = {}, c10 = {}, c11 = {};

  for (int k0 = 0; k0 < DMODEL; k0 += BK) {
#pragma unroll
    for (int rr = 0; rr < 4; ++rr) {   // A tile 128x32 fp32 -> bf16
      const int r = lr + rr * 32;
      const f32x4 a = *(const f32x4*)(x + (size_t)(m0 + r) * DMODEL + k0 + lc);
      unsigned short* d = As + r * LDA + lc;
      d[0] = f2bf(a[0]); d[1] = f2bf(a[1]); d[2] = f2bf(a[2]); d[3] = f2bf(a[3]);
    }
#pragma unroll
    for (int rr = 0; rr < 2; ++rr) {   // B tile = 64 rows of W (K-contiguous)
      const int r = lr + rr * 32;
      const f32x4 b = *(const f32x4*)(W + (size_t)(n0 + r) * DMODEL + k0 + lc);
      unsigned short* d = Bs + r * LDB + lc;
      d[0] = f2bf(b[0]); d[1] = f2bf(b[1]); d[2] = f2bf(b[2]); d[3] = f2bf(b[3]);
    }
    if (k0 + BK < DMODEL)
      __builtin_prefetch(x + (size_t)(m0 + lr) * DMODEL + k0 + BK + lc, 0, 1);
    __syncthreads();
    const bf16x16 a0 = frag_a(As, LDA, wm + 0,  lane);
    const bf16x16 a1 = frag_a(As, LDA, wm + 16, lane);
    const bf16x16 b0 = frag_b(Bs, LDB, wn + 0,  0, lane);
    const bf16x16 b1 = frag_b(Bs, LDB, wn + 16, 0, lane);
    c00 = wmma_bf16(a0, b0, c00);
    c01 = wmma_bf16(a0, b1, c01);
    c10 = wmma_bf16(a1, b0, c10);
    c11 = wmma_bf16(a1, b1, c11);
    __syncthreads();
  }

  const int ln = lane & 15;
  const int lh = (lane >> 4) * 8;
  f32x8 acc[2][2] = { { c00, c01 }, { c10, c11 } };
#pragma unroll
  for (int mi = 0; mi < 2; ++mi)
#pragma unroll
    for (int ni = 0; ni < 2; ++ni) {
      const int j = n0 + wn + ni * 16 + ln;
      const float bj = bias[j];
      const int h = j >> 6, dh = j & 63;
#pragma unroll
      for (int r = 0; r < 8; ++r) {
        const int m = m0 + wm + mi * 16 + r + lh;
        const int b = m >> 12, n = m & (NSEQ - 1);
        out[(((size_t)b * NHEAD + h) * NSEQ + n) * DHEAD + dh] = f2bf(acc[mi][ni][r] + bj);
      }
    }
}

// ================= Kernel 2: feature softmax on projected Q (in place) ======
__global__ __launch_bounds__(256)
void qsoftmax_kernel(unsigned short* __restrict__ qs)
{
  const size_t row = (size_t)blockIdx.x * 256 + threadIdx.x;  // < B*H*N
  unsigned short* p = qs + row * DHEAD;
  float v[DHEAD];
#pragma unroll
  for (int i = 0; i < 8; ++i) {
    const u16x8 h = *(const u16x8*)(p + i * 8);
#pragma unroll
    for (int j = 0; j < 8; ++j) v[i * 8 + j] = bf2f(h[j]);
  }
  float mx = v[0];
#pragma unroll
  for (int i = 1; i < DHEAD; ++i) mx = fmaxf(mx, v[i]);
  float s = 0.f;
#pragma unroll
  for (int i = 0; i < DHEAD; ++i) { v[i] = __expf(v[i] - mx); s += v[i]; }
  const float inv = INV_QTR / s;
#pragma unroll
  for (int i = 0; i < 8; ++i) {
    u16x8 h;
#pragma unroll
    for (int j = 0; j < 8; ++j) h[j] = f2bf(v[i * 8 + j] * inv);
    *(u16x8*)(p + i * 8) = h;
  }
}

// ================= Kernel 3: ctx partial = kp^T @ vp over an N chunk ========
// ctxp[chunk][bh][d][e] = sum_{n in chunk} kp[bh,n,d] * vp[bh,n,e]
__global__ __launch_bounds__(256)
void ctx_partial_kernel(const unsigned short* __restrict__ kp,
                        const unsigned short* __restrict__ vp,
                        float* __restrict__ ctxp)
{
  constexpr int LDT = 40;
  __shared__ unsigned short kT[DHEAD * LDT];  // [d][n] transposed tile
  __shared__ unsigned short vT[DHEAD * LDT];  // [e][n] transposed tile
  const int bh = blockIdx.x;
  const int chunk = blockIdx.y;
  const int tid = threadIdx.x, lane = tid & 31, w = tid >> 5;
  const int dt = (w & 3) * 16;
  const int et = (w >> 2) * 32;
  const int tn = tid >> 3;            // 0..31 : n row in tile
  const int td = (tid & 7) * 8;       // d/e block of 8
  const size_t base = (size_t)bh * NSEQ * DHEAD;
  f32x8 c0 = {}, c1 = {};

  for (int it = 0; it < 16; ++it) {
    const int nb = chunk * 512 + it * 32;
    const u16x8 hk = *(const u16x8*)(kp + base + (size_t)(nb + tn) * DHEAD + td);
    const u16x8 hv = *(const u16x8*)(vp + base + (size_t)(nb + tn) * DHEAD + td);
#pragma unroll
    for (int j = 0; j < 8; ++j) {     // transpose into LDS
      kT[(td + j) * LDT + tn] = hk[j];
      vT[(td + j) * LDT + tn] = hv[j];
    }
    __syncthreads();
    const bf16x16 a  = frag_a(kT, LDT, dt, lane);         // A: 16(d) x 32(n)
    const bf16x16 b0 = frag_b(vT, LDT, et + 0,  0, lane); // B: 32(n) x 16(e)
    const bf16x16 b1 = frag_b(vT, LDT, et + 16, 0, lane);
    c0 = wmma_bf16(a, b0, c0);
    c1 = wmma_bf16(a, b1, c1);
    __syncthreads();
  }

  float* dst = ctxp + ((size_t)chunk * (BATCH * NHEAD) + bh) * (DHEAD * DHEAD);
  const int ln = lane & 15;
  const int lh = (lane >> 4) * 8;
#pragma unroll
  for (int r = 0; r < 8; ++r) {
    const int d = dt + r + lh;
    dst[d * DHEAD + et + ln]      = c0[r];
    dst[d * DHEAD + et + 16 + ln] = c1[r];
  }
}

// ================= Kernel 4: ctx column softmax (axis=-2) -> bf16 ===========
__global__ __launch_bounds__(64)
void ctx_softmax_kernel(const float* __restrict__ ctxp, unsigned short* __restrict__ ctxb)
{
  const int bh = blockIdx.x;
  const int e = threadIdx.x;
  float v[DHEAD];
#pragma unroll
  for (int d = 0; d < DHEAD; ++d) {
    float s = 0.f;
#pragma unroll
    for (int c = 0; c < 8; ++c)
      s += ctxp[((size_t)c * (BATCH * NHEAD) + bh) * (DHEAD * DHEAD) + d * DHEAD + e];
    v[d] = s;
  }
  float mx = v[0];
#pragma unroll
  for (int d = 1; d < DHEAD; ++d) mx = fmaxf(mx, v[d]);
  float s = 0.f;
#pragma unroll
  for (int d = 0; d < DHEAD; ++d) { v[d] = __expf(v[d] - mx); s += v[d]; }
  const float inv = INV_QTR / s;
#pragma unroll
  for (int d = 0; d < DHEAD; ++d)
    ctxb[(size_t)bh * (DHEAD * DHEAD) + d * DHEAD + e] = f2bf(v[d] * inv);
}

// ================= Kernel 5: att = qs @ ctx, merged-head bf16 out ===========
__global__ __launch_bounds__(256)
void att_gemm_kernel(const unsigned short* __restrict__ qs,
                     const unsigned short* __restrict__ ctxb,
                     unsigned short* __restrict__ att)
{
  constexpr int LDC = 72;
  __shared__ unsigned short cT[DHEAD * LDC];  // ctx^T: [e][d]
  const int bh = blockIdx.x;
  const int m0 = blockIdx.y * 256;
  const int b = bh >> 4, h = bh & 15;
  const int tid = threadIdx.x, lane = tid & 31, w = tid >> 5;
  {
    const int d = tid >> 2;
    const int e0 = (tid & 3) * 16;
    const unsigned short* s = ctxb + (size_t)bh * (DHEAD * DHEAD) + d * DHEAD + e0;
    const u16x8 lo = *(const u16x8*)s;
    const u16x8 hi = *(const u16x8*)(s + 8);
#pragma unroll
    for (int j = 0; j < 8; ++j) {
      cT[(e0 + j) * LDC + d] = lo[j];
      cT[(e0 + 8 + j) * LDC + d] = hi[j];
    }
  }
  __syncthreads();

  const size_t qbase = (size_t)bh * NSEQ * DHEAD;
  const int mw = m0 + w * 32;
  f32x8 acc[2][4] = {};
#pragma unroll
  for (int k = 0; k < DHEAD; k += 32) {
    // A fragments straight from global (rows are K-contiguous, 64 bf16)
    const int kofs = (lane & 16) ? 8 : 0;
    const unsigned short* p0 = qs + qbase + (size_t)(mw + (lane & 15)) * DHEAD + k + kofs;
    const bf16x16 a0 = pack_frag(*(const u16x8*)p0, *(const u16x8*)(p0 + 16));
    const unsigned short* p1 = p0 + 16 * DHEAD;
    const bf16x16 a1 = pack_frag(*(const u16x8*)p1, *(const u16x8*)(p1 + 16));
#pragma unroll
    for (int ni = 0; ni < 4; ++ni) {
      const bf16x16 bf = frag_b(cT, LDC, ni * 16, k, lane);
      acc[0][ni] = wmma_bf16(a0, bf, acc[0][ni]);
      acc[1][ni] = wmma_bf16(a1, bf, acc[1][ni]);
    }
  }

  const int ln = lane & 15;
  const int lh = (lane >> 4) * 8;
#pragma unroll
  for (int mi = 0; mi < 2; ++mi)
#pragma unroll
    for (int ni = 0; ni < 4; ++ni)
#pragma unroll
      for (int r = 0; r < 8; ++r) {
        const int n = mw + mi * 16 + r + lh;
        const int e = ni * 16 + ln;
        att[((size_t)b * NSEQ + n) * DMODEL + h * DHEAD + e] = f2bf(acc[mi][ni][r]);
      }
}

// ================= Kernel 6: output GEMM (bf16 in, fp32 out, no bias) =======
__global__ __launch_bounds__(256)
void out_gemm_kernel(const unsigned short* __restrict__ x, const float* __restrict__ W,
                     float* __restrict__ out)
{
  constexpr int BM = 128, BN = 64, BK = 32, LDA = 40, LDB = 40;
  __shared__ unsigned short As[BM * LDA];
  __shared__ unsigned short Bs[BN * LDB];
  const int m0 = blockIdx.x * BM;
  const int n0 = blockIdx.y * BN;
  const int tid = threadIdx.x;
  const int lane = tid & 31;
  const int w = tid >> 5;
  const int wm = (w & 3) * 32;
  const int wn = (w >> 2) * 32;
  const int ar = tid >> 1;            // 0..127
  const int ac = (tid & 1) * 16;      // 0 or 16
  const int br = tid >> 3;            // 0..31
  const int bc = (tid & 7) * 4;

  f32x8 c00 = {}, c01 = {}, c10 = {}, c11 = {};

  for (int k0 = 0; k0 < DMODEL; k0 += BK) {
    {
      // A tile is a raw bf16 copy: use the CDNA5 async Global->LDS engine
      // (ASYNCcnt) when the toolchain exposes it; the imm offset applies to
      // both the global and LDS side per the ISA address formula.
      const unsigned short* s = x + (size_t)(m0 + ar) * DMODEL + k0 + ac;
      unsigned short* d = As + ar * LDA + ac;
#if USE_ASYNC_LDS
      __builtin_amdgcn_global_load_async_to_lds_b128(
          (as1_i32x4*)s, (as3_i32x4*)d, 0, 0);
      __builtin_amdgcn_global_load_async_to_lds_b128(
          (as1_i32x4*)s, (as3_i32x4*)d, 16, 0);
#else
      *(u16x8*)(d)     = *(const u16x8*)s;
      *(u16x8*)(d + 8) = *(const u16x8*)(s + 8);
#endif
    }
#pragma unroll
    for (int rr = 0; rr < 2; ++rr) {
      const int r = br + rr * 32;
      const f32x4 bv = *(const f32x4*)(W + (size_t)(n0 + r) * DMODEL + k0 + bc);
      unsigned short* d = Bs + r * LDB + bc;
      d[0] = f2bf(bv[0]); d[1] = f2bf(bv[1]); d[2] = f2bf(bv[2]); d[3] = f2bf(bv[3]);
    }
#if USE_ASYNC_LDS
    asm volatile("s_wait_asynccnt 0x0" ::: "memory");
#endif
    __syncthreads();
    const bf16x16 a0 = frag_a(As, LDA, wm + 0,  lane);
    const bf16x16 a1 = frag_a(As, LDA, wm + 16, lane);
    const bf16x16 b0 = frag_b(Bs, LDB, wn + 0,  0, lane);
    const bf16x16 b1 = frag_b(Bs, LDB, wn + 16, 0, lane);
    c00 = wmma_bf16(a0, b0, c00);
    c01 = wmma_bf16(a0, b1, c01);
    c10 = wmma_bf16(a1, b0, c10);
    c11 = wmma_bf16(a1, b1, c11);
    __syncthreads();
  }

  const int ln = lane & 15;
  const int lh = (lane >> 4) * 8;
  f32x8 acc[2][2] = { { c00, c01 }, { c10, c11 } };
#pragma unroll
  for (int mi = 0; mi < 2; ++mi)
#pragma unroll
    for (int ni = 0; ni < 2; ++ni) {
      const int j = n0 + wn + ni * 16 + ln;
#pragma unroll
      for (int r = 0; r < 8; ++r) {
        const int m = m0 + wm + mi * 16 + r + lh;
        out[(size_t)m * DMODEL + j] = acc[mi][ni][r];
      }
    }
}

// =============================== launch =====================================
extern "C" void kernel_launch(void* const* d_in, const int* in_sizes, int n_in,
                              void* d_out, int out_size, void* d_ws, size_t ws_size,
                              hipStream_t stream) {
  const float* q  = (const float*)d_in[0];
  const float* k  = (const float*)d_in[1];
  const float* v  = (const float*)d_in[2];
  const float* Wq = (const float*)d_in[3];
  const float* bq = (const float*)d_in[4];
  const float* Wk = (const float*)d_in[5];
  const float* bk = (const float*)d_in[6];
  const float* Wv = (const float*)d_in[7];
  const float* bv = (const float*)d_in[8];
  const float* Wo = (const float*)d_in[9];
  float* out = (float*)d_out;

  constexpr size_t PROJ = (size_t)BATCH * NHEAD * NSEQ * DHEAD;   // 16.78M elems
  unsigned short* qsw  = (unsigned short*)d_ws;                   // bf16 [bh][n][dh]
  unsigned short* kpw  = qsw + PROJ;
  unsigned short* vpw  = kpw + PROJ;
  float*          ctxp = (float*)(vpw + PROJ);                    // [8][bh][64][64]
  unsigned short* ctxb = (unsigned short*)(ctxp + (size_t)8 * BATCH * NHEAD * DHEAD * DHEAD);
  unsigned short* attw = kpw;  // kp is dead after ctx_partial; reuse for merged att

  const dim3 gproj(128, 16);  // 16384/128 x 1024/64
  proj_gemm_kernel<<<gproj, 256, 0, stream>>>(q, Wq, bq, qsw);
  proj_gemm_kernel<<<gproj, 256, 0, stream>>>(k, Wk, bk, kpw);
  proj_gemm_kernel<<<gproj, 256, 0, stream>>>(v, Wv, bv, vpw);
  qsoftmax_kernel<<<(BATCH * NHEAD * NSEQ) / 256, 256, 0, stream>>>(qsw);
  ctx_partial_kernel<<<dim3(BATCH * NHEAD, 8), 256, 0, stream>>>(kpw, vpw, ctxp);
  ctx_softmax_kernel<<<BATCH * NHEAD, 64, 0, stream>>>(ctxp, ctxb);
  att_gemm_kernel<<<dim3(BATCH * NHEAD, 16), 256, 0, stream>>>(qsw, ctxb, attw);
  out_gemm_kernel<<<gproj, 256, 0, stream>>>(attw, Wo, out);
}
